// MultiHierarchicalGRU_11690900980383
// MI455X (gfx1250) — matrix operand order repaired and checked by reading, not compiled, not run on hardware
//
#include <hip/hip_runtime.h>

typedef __bf16 bf16_t;
typedef __bf16 v8bf  __attribute__((ext_vector_type(8)));
typedef __bf16 v16bf __attribute__((ext_vector_type(16)));
typedef float  v8f   __attribute__((ext_vector_type(8)));

#define GRU_H   256
#define GRU_G3  768
#define EPAD    224   // 200 padded up to 7*32

__device__ __forceinline__ float sigm(float x) { return 1.f / (1.f + __expf(-x)); }

// ---------------------------------------------------------------------------
// Embedding gather -> bf16, time-major: X1[t][n][e], t<32, n<4096, e<EPAD
// ---------------------------------------------------------------------------
__global__ void embed_kernel(const int* __restrict__ toks,
                             const float* __restrict__ emb,
                             bf16_t* __restrict__ X1) {
  long long idx = (long long)blockIdx.x * 256 + threadIdx.x;
  const long long total = 32LL * 4096 * EPAD;
  if (idx >= total) return;
  int e = (int)(idx % EPAD);
  long long rem = idx / EPAD;
  int n1 = (int)(rem % 4096);
  int t  = (int)(rem / 4096);
  int tok = toks[n1 * 32 + t];
  float v = (e < 200) ? emb[(long long)tok * 200 + e] : 0.f;
  X1[idx] = (bf16_t)v;
}

// ---------------------------------------------------------------------------
// fp32 -> bf16 weight convert with K padding: dst[rows][kpad]
// ---------------------------------------------------------------------------
__global__ void convw_kernel(const float* __restrict__ src, bf16_t* __restrict__ dst,
                             int rows, int kin, int kpad) {
  int idx = blockIdx.x * 256 + threadIdx.x;
  if (idx >= rows * kpad) return;
  int r = idx / kpad, k = idx % kpad;
  dst[idx] = (k < kin) ? (bf16_t)src[r * kin + k] : (bf16_t)0.f;
}

__global__ void zero_kernel(unsigned int* __restrict__ p, int n) {
  int idx = blockIdx.x * 256 + threadIdx.x;
  if (idx < n) p[idx] = 0u;
}

// ---------------------------------------------------------------------------
// Fused GRU step. One workgroup (8 waves, 256 thr) owns a 16-row M-tile:
//   gx = x_t[16,K1] * Wih[768,K1]^T      (WMMA, 6 accs of 16x16 per wave)
//   gh = h  [16,256] * Whh[768,256]^T    (WMMA)
//   LDS planes: 0: gx_r+gh_r, 1: gx_z+gh_z, 2: gx_n, 3: gh_n   (16x256 f32 each)
//   gate:  r=sig(.+b), z=sig(.+b), n=tanh(xn+bxn + r*(hn+bhn))
//          h' = (1-z)*n + z*h   -> Hf (f32) and Hb (bf16, next step's A)
// Safe: this block reads only its OWN 16 rows of Hb before overwriting them.
// ---------------------------------------------------------------------------
__global__ void gru_step_kernel(const bf16_t* __restrict__ X, int K1,
                                const bf16_t* __restrict__ Wih,
                                const bf16_t* __restrict__ Whh,
                                const float* __restrict__ bih,
                                const float* __restrict__ bhh,
                                float* __restrict__ Hf,
                                bf16_t* __restrict__ Hb) {
  __shared__ float ls[4][16][GRU_H];   // 64 KB

  const int lane = threadIdx.x & 31;
  const int w    = threadIdx.x >> 5;   // wave 0..7 -> columns [96w, 96w+96)
  const int h    = lane >> 4;          // lane half selects K sub-block
  const int ml   = lane & 15;          // A: row, B: column
  const int m0   = blockIdx.x << 4;

  v8f agx[6] = {};
  v8f agh[6] = {};

  // ---- gx GEMM: A = X rows m0..m0+15 (row stride K1), B = Wih (ldb = K1)
  {
    const bf16_t* Arow = X + (size_t)(m0 + ml) * K1 + 16 * h;
    const bf16_t* Bb   = Wih + (size_t)(96 * w + ml) * K1 + 16 * h;
    for (int k0 = 0; k0 < K1; k0 += 32) {
      v8bf alo = *(const v8bf*)(Arow + k0);
      v8bf ahi = *(const v8bf*)(Arow + k0 + 8);
      v16bf a = __builtin_shufflevector(alo, ahi,
                                        0,1,2,3,4,5,6,7,8,9,10,11,12,13,14,15);
#pragma unroll
      for (int c = 0; c < 6; ++c) {
        v16bf b = *(const v16bf*)(Bb + (size_t)(16 * c) * K1 + k0);
        agx[c] = __builtin_amdgcn_wmma_f32_16x16x32_bf16(false, a, false, b,
                                                         (short)0, agx[c],
                                                         false, false);
      }
    }
  }
  // ---- gh GEMM: A = Hb rows m0..m0+15 (ld 256), B = Whh (ldb = 256)
  {
    const bf16_t* Arow = Hb + (size_t)(m0 + ml) * GRU_H + 16 * h;
    const bf16_t* Bb   = Whh + (size_t)(96 * w + ml) * GRU_H + 16 * h;
    for (int k0 = 0; k0 < GRU_H; k0 += 32) {
      v8bf alo = *(const v8bf*)(Arow + k0);
      v8bf ahi = *(const v8bf*)(Arow + k0 + 8);
      v16bf a = __builtin_shufflevector(alo, ahi,
                                        0,1,2,3,4,5,6,7,8,9,10,11,12,13,14,15);
#pragma unroll
      for (int c = 0; c < 6; ++c) {
        v16bf b = *(const v16bf*)(Bb + (size_t)(16 * c) * GRU_H + k0);
        agh[c] = __builtin_amdgcn_wmma_f32_16x16x32_bf16(false, a, false, b,
                                                         (short)0, agh[c],
                                                         false, false);
      }
    }
  }

  // ---- spill to LDS (D layout: lane col = ml, VGPR r -> row r + 8h)
#pragma unroll
  for (int c = 0; c < 6; ++c) {
    int n0  = 96 * w + 16 * c;       // 16-aligned, never straddles a 256 region
    int reg = n0 >> 8;               // 0:r, 1:z, 2:n
    int jc  = (n0 & 255) + ml;
    int rb  = 8 * h;
    if (reg < 2) {
#pragma unroll
      for (int r = 0; r < 8; ++r)
        ls[reg][rb + r][jc] = agx[c][r] + agh[c][r];
    } else {
#pragma unroll
      for (int r = 0; r < 8; ++r) {
        ls[2][rb + r][jc] = agx[c][r];
        ls[3][rb + r][jc] = agh[c][r];
      }
    }
  }
  __syncthreads();

  // ---- gate math: thread j handles column j for all 16 rows
  {
    const int j = threadIdx.x;
    const float bxr = bih[j],            bhr = bhh[j];
    const float bxz = bih[GRU_H + j],    bhz = bhh[GRU_H + j];
    const float bxn = bih[2*GRU_H + j],  bhn = bhh[2*GRU_H + j];
#pragma unroll 4
    for (int i = 0; i < 16; ++i) {
      float r  = sigm(ls[0][i][j] + bxr + bhr);
      float z  = sigm(ls[1][i][j] + bxz + bhz);
      float nn = tanhf(ls[2][i][j] + bxn + r * (ls[3][i][j] + bhn));
      size_t idx = (size_t)(m0 + i) * GRU_H + j;
      float hv = Hf[idx];
      float hn = (1.f - z) * nn + z * hv;
      Hf[idx] = hn;
      Hb[idx] = (bf16_t)hn;
    }
  }
}

// ---------------------------------------------------------------------------
// Rearrange level output into next level's time-major input:
// dst[t][n][c] = src[n*T + t][c]   (src: [Nseq*T, 256] bf16)
// ---------------------------------------------------------------------------
__global__ void rearr_kernel(const bf16_t* __restrict__ src, bf16_t* __restrict__ dst,
                             int Nseq, int T) {
  int idx = blockIdx.x * 256 + threadIdx.x;
  if (idx >= Nseq * T * GRU_H) return;
  int c = idx & 255;
  int rem = idx >> 8;
  int n = rem % Nseq;
  int t = rem / Nseq;
  dst[idx] = src[((size_t)(n * T + t) << 8) + c];
}

// ---------------------------------------------------------------------------
// MLP head: out[row] = SELU(x @ W1^T + b1) @ W2^T + b2.  W1:[128,256] W2:[1,128]
// One block (128 threads) per row.
// ---------------------------------------------------------------------------
__global__ void mlp_kernel(const float* __restrict__ X,
                           const float* __restrict__ W1, const float* __restrict__ b1,
                           const float* __restrict__ W2, const float* __restrict__ b2,
                           float* __restrict__ out) {
  __shared__ float s[128];
  int row = blockIdx.x;
  int j = threadIdx.x;
  const float* x = X  + (size_t)row * GRU_H;
  const float* wv = W1 + (size_t)j * GRU_H;
  float d = 0.f;
  for (int k = 0; k < GRU_H; ++k) d += x[k] * wv[k];
  d += b1[j];
  const float kScale = 1.0507009873554804934f;
  const float kAlpha = 1.6732632423543772848f;
  float hid = d > 0.f ? kScale * d : kScale * kAlpha * (__expf(d) - 1.f);
  s[j] = hid * W2[j];
  __syncthreads();
  for (int st = 64; st > 0; st >>= 1) {
    if (j < st) s[j] += s[j + st];
    __syncthreads();
  }
  if (j == 0) out[row] = s[0] + b2[0];
}

// ---------------------------------------------------------------------------
extern "C" void kernel_launch(void* const* d_in, const int* in_sizes, int n_in,
                              void* d_out, int out_size, void* d_ws, size_t ws_size,
                              hipStream_t stream) {
  const int*   toks  = (const int*)d_in[0];
  const float* emb   = (const float*)d_in[1];
  const float* wWih  = (const float*)d_in[2];
  const float* wWhh  = (const float*)d_in[3];
  const float* wbih  = (const float*)d_in[4];
  const float* wbhh  = (const float*)d_in[5];
  const float* sWih  = (const float*)d_in[6];
  const float* sWhh  = (const float*)d_in[7];
  const float* sbih  = (const float*)d_in[8];
  const float* sbhh  = (const float*)d_in[9];
  const float* rWih  = (const float*)d_in[10];
  const float* rWhh  = (const float*)d_in[11];
  const float* rbih  = (const float*)d_in[12];
  const float* rbhh  = (const float*)d_in[13];
  const float* rfcW1 = (const float*)d_in[14];
  const float* rfcb1 = (const float*)d_in[15];
  const float* rfcW2 = (const float*)d_in[16];
  const float* rfcb2 = (const float*)d_in[17];
  const float* pfcW1 = (const float*)d_in[18];
  const float* pfcb1 = (const float*)d_in[19];
  const float* pfcW2 = (const float*)d_in[20];
  const float* pfcb2 = (const float*)d_in[21];
  float* out = (float*)d_out;   // [0..15]=b_stars, [16..271]=r_stars

  char* ws = (char*)d_ws;
  size_t off = 0;
  auto alloc = [&](size_t bytes) -> char* {
    char* p = ws + off;
    off += (bytes + 255) & ~(size_t)255;
    return p;
  };
  bf16_t* X1   = (bf16_t*)alloc(32ull * 4096 * EPAD * 2);   // word inputs (time-major)
  bf16_t* X2   = (bf16_t*)alloc(16ull * 256 * GRU_H * 2);   // sentence inputs
  bf16_t* X3   = (bf16_t*)alloc(16ull * 16 * GRU_H * 2);    // review inputs
  float*  Hf   = (float*) alloc(4096ull * GRU_H * 4);       // fp32 hidden (max M)
  bf16_t* Hb   = (bf16_t*)alloc(4096ull * GRU_H * 2);       // bf16 hidden copy
  bf16_t* WwIh = (bf16_t*)alloc(768ull * EPAD * 2);
  bf16_t* WwHh = (bf16_t*)alloc(768ull * GRU_H * 2);
  bf16_t* WsIh = (bf16_t*)alloc(768ull * GRU_H * 2);
  bf16_t* WsHh = (bf16_t*)alloc(768ull * GRU_H * 2);
  bf16_t* WrIh = (bf16_t*)alloc(768ull * GRU_H * 2);
  bf16_t* WrHh = (bf16_t*)alloc(768ull * GRU_H * 2);
  (void)ws_size; (void)in_sizes; (void)n_in; (void)out_size;

  // --- weight conversion to padded bf16
  convw_kernel<<<(768 * EPAD  + 255) / 256, 256, 0, stream>>>(wWih, WwIh, 768, 200, EPAD);
  convw_kernel<<<(768 * GRU_H + 255) / 256, 256, 0, stream>>>(wWhh, WwHh, 768, 256, GRU_H);
  convw_kernel<<<(768 * GRU_H + 255) / 256, 256, 0, stream>>>(sWih, WsIh, 768, 256, GRU_H);
  convw_kernel<<<(768 * GRU_H + 255) / 256, 256, 0, stream>>>(sWhh, WsHh, 768, 256, GRU_H);
  convw_kernel<<<(768 * GRU_H + 255) / 256, 256, 0, stream>>>(rWih, WrIh, 768, 256, GRU_H);
  convw_kernel<<<(768 * GRU_H + 255) / 256, 256, 0, stream>>>(rWhh, WrHh, 768, 256, GRU_H);

  // --- embedding gather
  embed_kernel<<<(int)((32ull * 4096 * EPAD + 255) / 256), 256, 0, stream>>>(toks, emb, X1);

  // ============ word-level GRU: M=4096, 32 fused steps ============
  zero_kernel<<<(4096 * 256 + 255) / 256, 256, 0, stream>>>((unsigned int*)Hf, 4096 * 256);
  zero_kernel<<<(4096 * 128 + 255) / 256, 256, 0, stream>>>((unsigned int*)Hb, 4096 * 128);
  for (int t = 0; t < 32; ++t)
    gru_step_kernel<<<4096 / 16, 256, 0, stream>>>(X1 + (size_t)t * 4096 * EPAD, EPAD,
                                                   WwIh, WwHh, wbih, wbhh, Hf, Hb);
  // sent_h (bf16 in Hb, rows n2*16+t) -> X2 time-major
  rearr_kernel<<<(256 * 16 * 256 + 255) / 256, 256, 0, stream>>>(Hb, X2, 256, 16);

  // ============ sentence-level GRU: M=256, 16 fused steps ============
  zero_kernel<<<(256 * 256 + 255) / 256, 256, 0, stream>>>((unsigned int*)Hf, 256 * 256);
  zero_kernel<<<(256 * 128 + 255) / 256, 256, 0, stream>>>((unsigned int*)Hb, 256 * 128);
  for (int t = 0; t < 16; ++t)
    gru_step_kernel<<<256 / 16, 256, 0, stream>>>(X2 + (size_t)t * 256 * GRU_H, GRU_H,
                                                  WsIh, WsHh, sbih, sbhh, Hf, Hb);
  // r_stars MLP on rev_h (fp32 in Hf) BEFORE Hf is reused
  mlp_kernel<<<256, 128, 0, stream>>>(Hf, rfcW1, rfcb1, rfcW2, rfcb2, out + 16);
  // rev_h (bf16 in Hb, rows b*16+r) -> X3 time-major
  rearr_kernel<<<(16 * 16 * 256 + 255) / 256, 256, 0, stream>>>(Hb, X3, 16, 16);

  // ============ review-level GRU: M=16, 16 fused steps ============
  zero_kernel<<<(16 * 256 + 255) / 256, 256, 0, stream>>>((unsigned int*)Hf, 16 * 256);
  zero_kernel<<<(16 * 128 + 255) / 256, 256, 0, stream>>>((unsigned int*)Hb, 16 * 128);
  for (int t = 0; t < 16; ++t)
    gru_step_kernel<<<1, 256, 0, stream>>>(X3 + (size_t)t * 16 * GRU_H, GRU_H,
                                           WrIh, WrHh, rbih, rbhh, Hf, Hb);
  // b_stars MLP on biz_h (fp32 in Hf)
  mlp_kernel<<<16, 128, 0, stream>>>(Hf, pfcW1, pfcb1, pfcW2, pfcb2, out);
}
